// DABlock_90237262889329
// MI455X (gfx1250) — compile-verified, matrix operand order checked by reading, not run on hardware
//
#include <hip/hip_runtime.h>

// ---------------------------------------------------------------------------
// Types for CDNA5 WMMA (wave32): bf16 16x16x32, f32 accumulate
// ---------------------------------------------------------------------------
typedef __attribute__((ext_vector_type(16))) __bf16         v16bf;
typedef __attribute__((ext_vector_type(8)))  float          v8f;
typedef __attribute__((ext_vector_type(16))) unsigned short v16u;
typedef __attribute__((ext_vector_type(8)))  unsigned short u16x8;
typedef __attribute__((ext_vector_type(4)))  unsigned int   u32x4;
typedef __attribute__((ext_vector_type(8)))  int            i32x8;
typedef __attribute__((ext_vector_type(4)))  int            i32x4;

#if defined(__has_builtin)
#if __has_builtin(__builtin_amdgcn_tensor_load_to_lds)
#define HAVE_TDM 1
#else
#define HAVE_TDM 0
#endif
#else
#define HAVE_TDM 0
#endif

union FragU { v16u v; u16x8 h[2]; };

static __device__ __forceinline__ unsigned short f2bf(float x) {
  unsigned int u = __builtin_bit_cast(unsigned int, x);
  u += 0x7fffu + ((u >> 16) & 1u);   // round-to-nearest-even
  return (unsigned short)(u >> 16);
}
static __device__ __forceinline__ float bf2f(unsigned short h) {
  unsigned int u = ((unsigned int)h) << 16;
  return __builtin_bit_cast(float, u);
}

// K-contiguous fragment: element e<8 -> k=base+e, e>=8 -> k=base+16+(e-8);
// caller passes row pointer already offset by base. Reads [0..7],[16..23].
static __device__ __forceinline__ v16bf load_frag(const unsigned short* p) {
  FragU f;
  f.h[0] = *reinterpret_cast<const u16x8*>(p);
  f.h[1] = *reinterpret_cast<const u16x8*>(p + 16);
  return __builtin_bit_cast(v16bf, f.v);
}

static __device__ __forceinline__ v8f wmma_bf16(v16bf a, v16bf b, v8f c) {
  return __builtin_amdgcn_wmma_f32_16x16x32_bf16(false, a, false, b,
                                                 (short)0, c, false, false);
}

// ---------------------------------------------------------------------------
// Tensor Data Mover: 2-D tile load (global -> LDS). D# per CDNA5 ISA ch.8.
//   tileX contiguous elements per line, tileY lines, line pitch strideElems.
//   data_size = 2 bytes. Lines land back-to-back in LDS (pitch = tileX).
// 6-arg builtin on this toolchain; aux/extended groups zeroed for 2-D tiles.
// ---------------------------------------------------------------------------
static __device__ __forceinline__ void tdm_load_2d(unsigned ldsAddr,
                                                   const void* gaddr,
                                                   unsigned tileX, unsigned tileY,
                                                   unsigned long long strideElems) {
#if HAVE_TDM
  unsigned long long ga = (unsigned long long)(size_t)gaddr;
  u32x4 g0;
  g0[0] = 1u;                                        // count=1, user mode
  g0[1] = ldsAddr;                                   // lds_addr (bytes)
  g0[2] = (unsigned)(ga & 0xffffffffu);              // global_addr[31:0]
  g0[3] = (unsigned)((ga >> 32) & 0x01ffffffu)       // global_addr[56:32]
        | (2u << 30);                                // type=2 ("image")
  const unsigned tensorX = 1u << 24, tensorY = 1u << 24;  // no OOB clamping
  i32x8 g1;
  g1[0] = (int)(1u << 16);                           // data_size = 2 bytes
  g1[1] = (int)((tensorX & 0xffffu) << 16);                       // tdim0 lo
  g1[2] = (int)(((tensorX >> 16) & 0xffffu) | ((tensorY & 0xffffu) << 16));
  g1[3] = (int)(((tensorY >> 16) & 0xffffu) | ((tileX & 0xffffu) << 16));
  g1[4] = (int)(tileY & 0xffffu);                                 // tile_dim1
  g1[5] = (int)(unsigned)(strideElems & 0xffffffffu);             // stride lo
  g1[6] = (int)(unsigned)((strideElems >> 32) & 0xffffu);         // stride hi
  g1[7] = 0;
  i32x4 z4;
  z4[0] = 0; z4[1] = 0; z4[2] = 0; z4[3] = 0;
  i32x8 z8;
  z8[0] = 0; z8[1] = 0; z8[2] = 0; z8[3] = 0;
  z8[4] = 0; z8[5] = 0; z8[6] = 0; z8[7] = 0;
  __builtin_amdgcn_tensor_load_to_lds(g0, g1, z4, z4, z8, 0);
#else
  (void)ldsAddr; (void)gaddr; (void)tileX; (void)tileY; (void)strideElems;
#endif
}

static __device__ __forceinline__ void tdm_wait() {
#if HAVE_TDM
  __builtin_amdgcn_s_wait_tensorcnt(0);
#endif
}

// Fallback staging when no TDM builtin (keeps file compiling everywhere).
static __device__ __forceinline__ void stage2d_fallback(
    unsigned short* dst, const unsigned short* src, int tileX, int tileY,
    long long srcPitch, int tid, int nthr) {
#if !HAVE_TDM
  for (int i = tid; i < tileX * tileY; i += nthr) {
    int y = i / tileX, x = i - y * tileX;
    dst[i] = src[(long long)y * srcPitch + x];
  }
#else
  (void)dst; (void)src; (void)tileX; (void)tileY;
  (void)srcPitch; (void)tid; (void)nthr;
#endif
}

// ---------------------------------------------------------------------------
// Elementwise kernels
// ---------------------------------------------------------------------------
__global__ void cvt_bf16_kernel(const float* __restrict__ s,
                                unsigned short* __restrict__ d, size_t n) {
  size_t stride = (size_t)gridDim.x * blockDim.x;
  for (size_t i = (size_t)blockIdx.x * blockDim.x + threadIdx.x; i < n; i += stride)
    d[i] = f2bf(s[i]);
}

// f32 [B][C][N] -> bf16 [B][N][C]  (channel-minor for TDM K-contiguous tiles)
__global__ void transpose_cvt_kernel(const float* __restrict__ s,
                                     unsigned short* __restrict__ d,
                                     int C, int N, int B) {
  size_t total = (size_t)B * C * N;
  size_t stride = (size_t)gridDim.x * blockDim.x;
  for (size_t i = (size_t)blockIdx.x * blockDim.x + threadIdx.x; i < total; i += stride) {
    size_t cn = (size_t)C * N;
    int b = (int)(i / cn);
    size_t r = i - (size_t)b * cn;
    int c = (int)(r / N);
    int n = (int)(r - (size_t)c * N);
    d[((size_t)b * N + n) * C + c] = f2bf(s[i]);
  }
}

// [O,C,3,3] f32 -> [3,3,O,C] bf16 (dense (kh,kw) weight planes for TDM tiles)
__global__ void repack_w_kernel(const float* __restrict__ s,
                                unsigned short* __restrict__ d, int O, int C) {
  size_t n = (size_t)O * C * 9;
  size_t stride = (size_t)gridDim.x * blockDim.x;
  for (size_t i = (size_t)blockIdx.x * blockDim.x + threadIdx.x; i < n; i += stride) {
    int c = (int)(i % C);
    size_t t = i / C;
    int o = (int)(t % O);
    int k = (int)(t / O);
    int kh = k / 3, kw = k % 3;
    d[i] = f2bf(s[(((size_t)o * C + c) * 3 + kh) * 3 + kw]);
  }
}

// pad [O,C] f32 -> [Opad,C] bf16 (zero rows O..Opad-1)
__global__ void pad_cvt_kernel(const float* __restrict__ s,
                               unsigned short* __restrict__ d,
                               int O, int C, int Opad) {
  size_t n = (size_t)Opad * C;
  size_t stride = (size_t)gridDim.x * blockDim.x;
  for (size_t i = (size_t)blockIdx.x * blockDim.x + threadIdx.x; i < n; i += stride) {
    int o = (int)(i / C);
    d[i] = (o < O) ? f2bf(s[i]) : (unsigned short)0;
  }
}

__global__ void add_cvt_kernel(const unsigned short* __restrict__ a,
                               const unsigned short* __restrict__ b,
                               unsigned short* __restrict__ d, size_t n) {
  size_t stride = (size_t)gridDim.x * blockDim.x;
  for (size_t i = (size_t)blockIdx.x * blockDim.x + threadIdx.x; i < n; i += stride)
    d[i] = f2bf(bf2f(a[i]) + bf2f(b[i]));
}

// ---------------------------------------------------------------------------
// 3x3 conv (SAME) + BN + ReLU, implicit GEMM, TDM-staged K-contiguous tiles.
// Input N-major [B, 4096, Cin]; LDS: wsm[32 o][32 c], xs[66 pix][32 c].
// All fragments are contiguous-K -> ds_load_b128 pairs.
// outN: bf16 [B, 4096, Cout]; outC (optional): bf16 [B, Cout, 4096].
// Grid: (Cout/32, H=64, B); block 256 (8 waves). Weights [3][3][O][C].
// ---------------------------------------------------------------------------
__global__ void conv3x3_bn_relu_kernel(const unsigned short* __restrict__ xinN,
                                       const unsigned short* __restrict__ wgt,
                                       const float* __restrict__ gw,
                                       const float* __restrict__ gb,
                                       const float* __restrict__ gm,
                                       const float* __restrict__ gv,
                                       unsigned short* __restrict__ outN,
                                       unsigned short* __restrict__ outC,
                                       int Cin, int Cout) {
  __shared__ __align__(16) unsigned short smem[1024 + 66 * 32];
  unsigned short* wsm = smem;        // [o][c], pitch 32
  unsigned short* xs  = smem + 1024; // [pix(-1..64)][c], pitch 32
  const unsigned ldsBase = (unsigned)(size_t)(void*)smem;
  const unsigned wsmAddr = ldsBase;
  const unsigned xsAddr  = ldsBase + 2048u;

  const int tid  = threadIdx.x;
  const int lane = tid & 31;
  const int wv   = tid >> 5;
  const int mi = wv >> 2, ni = wv & 3;
  const int lidx = lane & 15;
  const int baseOff = (lane >> 4) << 3;
  const int oc0 = blockIdx.x * 32;
  const int h   = blockIdx.y;
  const int b   = blockIdx.z;

  v8f acc = {};
  for (int c0 = 0; c0 < Cin; c0 += 32) {
    for (int kh = 0; kh < 3; ++kh) {
      int hh = h + kh - 1;
      bool hok = (hh >= 0) && (hh < 64);
      __syncthreads();
      if (hok) {
        // 66 pixel lines starting at pixel hh*64 - 1 (w = -1 .. 64)
        const unsigned short* src =
            xinN + ((size_t)b * 4096 + (size_t)hh * 64 - 1) * Cin + c0;
        if (wv == 0) {
          tdm_load_2d(xsAddr, src, 32, 66, (unsigned long long)Cin);
          tdm_wait();
          // zero the W-halo lines (w = -1 and w = 64)
          xs[lane] = 0;
          xs[65 * 32 + lane] = 0;
        }
        // fallback path: interior lines + explicit halo zero (disjoint)
        stage2d_fallback(xs + 32, src + Cin, 32, 64, Cin, tid, 256);
#if !HAVE_TDM
        if (tid < 64) xs[(tid & 1) ? (65 * 32 + (tid >> 1)) : (tid >> 1)] = 0;
#endif
      } else {
        for (int i = tid; i < 66 * 32; i += 256) xs[i] = 0;
      }
      for (int kw = 0; kw < 3; ++kw) {
        if (kw) __syncthreads();
        const unsigned short* wsrc =
            wgt + (((size_t)(kh * 3 + kw) * Cout + oc0) * Cin + c0);
        if (wv == 0) {
          tdm_load_2d(wsmAddr, wsrc, 32, 32, (unsigned long long)Cin);
          tdm_wait();
        }
        stage2d_fallback(wsm, wsrc, 32, 32, Cin, tid, 256);
        __syncthreads();
        v16bf af = load_frag(&wsm[(mi * 16 + lidx) * 32 + baseOff]);
        v16bf bf = load_frag(&xs[(ni * 16 + lidx + kw) * 32 + baseOff]);
        acc = wmma_bf16(af, bf, acc);
      }
    }
  }
  // epilogue: BN + ReLU -> bf16 (N-major, optionally also C-major)
  const int w = ni * 16 + lidx;
  const int n = h * 64 + w;
  const int mbase = oc0 + mi * 16 + ((lane >> 4) << 3);
#pragma unroll
  for (int r = 0; r < 8; ++r) {
    int o = mbase + r;
    float sc = gw[o] * rsqrtf(gv[o] + 1e-5f);
    float val = (acc[r] - gm[o]) * sc + gb[o];
    val = fmaxf(val, 0.0f);
    unsigned short bv16 = f2bf(val);
    outN[((size_t)b * 4096 + n) * Cout + o] = bv16;
    if (outC) outC[((size_t)b * Cout + o) * 4096 + n] = bv16;
  }
}

// ---------------------------------------------------------------------------
// GEMM with TDM-staged K-contiguous tiles: out[o,n] = sum_c A[o,c]*BinN[n,c]
// BinN is N-major [B, 4096, Cin]. LDS: wsm[32 o][32 c], xs[64 n][32 c].
// mode 0: bf16 out C-major [B,Cout,4096] (+bias)
// mode 1: f32  out [B,Oreal,4096] (+bias), rows o<Oreal only
// mode 2: bf16 out N-major [B,4096,Cout] (+bias)
// mode 3: bf16 out N-major = gamma*acc + resN (no bias)   (CAM apply)
// Grid: (Cout/32, 64, B); block 256.
// ---------------------------------------------------------------------------
__global__ void gemm_ldsB_kernel(const unsigned short* __restrict__ A, size_t aBatch,
                                 const unsigned short* __restrict__ BinN,
                                 const float* __restrict__ bias,
                                 int Cin, int Cout, int Oreal, int mode,
                                 unsigned short* __restrict__ outBf,
                                 float* __restrict__ outF,
                                 const float* __restrict__ gammaPtr,
                                 const unsigned short* __restrict__ resN) {
  __shared__ __align__(16) unsigned short smem[1024 + 64 * 32];
  unsigned short* wsm = smem;        // [o][c], pitch 32
  unsigned short* xs  = smem + 1024; // [n][c], pitch 32
  const unsigned ldsBase = (unsigned)(size_t)(void*)smem;
  const unsigned wsmAddr = ldsBase;
  const unsigned xsAddr  = ldsBase + 2048u;

  const int tid  = threadIdx.x;
  const int lane = tid & 31;
  const int wv   = tid >> 5;
  const int mi = wv >> 2, ni = wv & 3;
  const int lidx = lane & 15;
  const int baseOff = (lane >> 4) << 3;
  const int oc0 = blockIdx.x * 32;
  const int n0  = blockIdx.y * 64;
  const int b   = blockIdx.z;

  v8f acc = {};
  for (int c0 = 0; c0 < Cin; c0 += 32) {
    __syncthreads();
    const unsigned short* bsrc = BinN + ((size_t)b * 4096 + n0) * Cin + c0;
    const unsigned short* asrc = A + (size_t)b * aBatch + (size_t)oc0 * Cin + c0;
    if (wv == 0) {
      tdm_load_2d(xsAddr, bsrc, 32, 64, (unsigned long long)Cin);
      tdm_load_2d(wsmAddr, asrc, 32, 32, (unsigned long long)Cin);
      tdm_wait();
    }
    stage2d_fallback(xs, bsrc, 32, 64, Cin, tid, 256);
    stage2d_fallback(wsm, asrc, 32, 32, Cin, tid, 256);
    __syncthreads();
    v16bf af = load_frag(&wsm[(mi * 16 + lidx) * 32 + baseOff]);
    v16bf bf = load_frag(&xs[(ni * 16 + lidx) * 32 + baseOff]);
    acc = wmma_bf16(af, bf, acc);
  }
  const int n = n0 + ni * 16 + lidx;
  const int mbase = oc0 + mi * 16 + ((lane >> 4) << 3);
#pragma unroll
  for (int r = 0; r < 8; ++r) {
    int o = mbase + r;
    float val = acc[r];
    if (mode == 3) {
      size_t idx = ((size_t)b * 4096 + n) * Cout + o;   // N-major
      val = gammaPtr[0] * val + bf2f(resN[idx]);
      outBf[idx] = f2bf(val);
    } else {
      if (bias) val += bias[o];
      if (mode == 0) {
        outBf[((size_t)b * Cout + o) * 4096 + n] = f2bf(val);
      } else if (mode == 1) {
        if (o < Oreal) outF[((size_t)b * Oreal + o) * 4096 + n] = val;
      } else { // mode 2: N-major
        outBf[((size_t)b * 4096 + n) * Cout + o] = f2bf(val);
      }
    }
  }
}

// ---------------------------------------------------------------------------
// Register-only GEMM: out[m,n] = sum_k A[m,k]*B[n,k] (both K-contiguous)
// mode 0: bf16 out; mode 1: f32 out; mode 2: bf16 out = gamma*acc + res
// outTrans: 0 -> idx = row*ldo + col; 1 -> idx = col*ldo + row
// Grid: (M/32, N/64, B); block 256.
// ---------------------------------------------------------------------------
__global__ void gemm_tt_kernel(const unsigned short* __restrict__ A, size_t aBatch, int lda,
                               const unsigned short* __restrict__ Bm, size_t bBatch, int ldb,
                               int K, int mode, int outTrans,
                               unsigned short* __restrict__ outBf,
                               float* __restrict__ outF,
                               size_t oBatch, int ldo,
                               const float* __restrict__ gammaPtr,
                               const unsigned short* __restrict__ resBf) {
  const int tid  = threadIdx.x;
  const int lane = tid & 31;
  const int wv   = tid >> 5;
  const int mi = wv >> 2, ni = wv & 3;
  const int lidx = lane & 15;
  const int baseOff = (lane >> 4) << 3;
  const int m0 = blockIdx.x * 32;
  const int n0 = blockIdx.y * 64;
  const int b  = blockIdx.z;

  const unsigned short* arow = A  + (size_t)b * aBatch + (size_t)(m0 + mi * 16 + lidx) * lda;
  const unsigned short* brow = Bm + (size_t)b * bBatch + (size_t)(n0 + ni * 16 + lidx) * ldb;

  v8f acc = {};
  for (int k0 = 0; k0 < K; k0 += 32) {
    if ((k0 & 511) == 0) {                       // global_prefetch_b8
      __builtin_prefetch(arow + k0 + 512, 0, 1);
      __builtin_prefetch(brow + k0 + 512, 0, 1);
    }
    v16bf af = load_frag(arow + k0 + baseOff);
    v16bf bf = load_frag(brow + k0 + baseOff);
    acc = wmma_bf16(af, bf, acc);
  }
  const int col = n0 + ni * 16 + lidx;
  const int rbase = m0 + mi * 16 + ((lane >> 4) << 3);
#pragma unroll
  for (int r = 0; r < 8; ++r) {
    int row = rbase + r;
    size_t idx = (size_t)b * oBatch +
                 (outTrans ? ((size_t)col * ldo + row) : ((size_t)row * ldo + col));
    if (mode == 0)      outBf[idx] = f2bf(acc[r]);
    else if (mode == 1) outF[idx]  = acc[r];
    else {
      float v = gammaPtr[0] * acc[r] + bf2f(resBf[idx]);
      outBf[idx] = f2bf(v);
    }
  }
}

// ---------------------------------------------------------------------------
// Row softmax (in place, bf16, length L). Grid = #rows, block 256.
// ---------------------------------------------------------------------------
__global__ void softmax_rows_kernel(unsigned short* __restrict__ attn, int L) {
  __shared__ float red[256];
  const int tid = threadIdx.x;
  unsigned short* p = attn + (size_t)blockIdx.x * L;
  float mx = -3.0e38f;
  for (int i = tid; i < L; i += 256) mx = fmaxf(mx, bf2f(p[i]));
  red[tid] = mx; __syncthreads();
  for (int s = 128; s > 0; s >>= 1) { if (tid < s) red[tid] = fmaxf(red[tid], red[tid + s]); __syncthreads(); }
  mx = red[0]; __syncthreads();
  float sum = 0.f;
  for (int i = tid; i < L; i += 256) sum += __expf(bf2f(p[i]) - mx);
  red[tid] = sum; __syncthreads();
  for (int s = 128; s > 0; s >>= 1) { if (tid < s) red[tid] += red[tid + s]; __syncthreads(); }
  float inv = 1.f / red[0];
  for (int i = tid; i < L; i += 256) p[i] = f2bf(__expf(bf2f(p[i]) - mx) * inv);
}

// CAM: softmax(rowmax - e) over d == exp(rowmin - e)/sum. L=512.
__global__ void cam_softmax_kernel(const float* __restrict__ energy,
                                   unsigned short* __restrict__ attnC) {
  __shared__ float red[256];
  const int tid = threadIdx.x;
  const float* e = energy + (size_t)blockIdx.x * 512;
  unsigned short* o = attnC + (size_t)blockIdx.x * 512;
  float mn = 3.0e38f;
  for (int i = tid; i < 512; i += 256) mn = fminf(mn, e[i]);
  red[tid] = mn; __syncthreads();
  for (int s = 128; s > 0; s >>= 1) { if (tid < s) red[tid] = fminf(red[tid], red[tid + s]); __syncthreads(); }
  mn = red[0]; __syncthreads();
  float sum = 0.f;
  for (int i = tid; i < 512; i += 256) sum += __expf(mn - e[i]);
  red[tid] = sum; __syncthreads();
  for (int s = 128; s > 0; s >>= 1) { if (tid < s) red[tid] += red[tid + s]; __syncthreads(); }
  float inv = 1.f / red[0];
  for (int i = tid; i < 512; i += 256) o[i] = f2bf(__expf(mn - e[i]) * inv);
}

// ---------------------------------------------------------------------------
// Host orchestration
// ---------------------------------------------------------------------------
extern "C" void kernel_launch(void* const* d_in, const int* in_sizes, int n_in,
                              void* d_out, int out_size, void* d_ws, size_t ws_size,
                              hipStream_t stream) {
  (void)in_sizes; (void)n_in; (void)out_size; (void)ws_size;
  const float* x    = (const float*)d_in[0];
  const float* w5a  = (const float*)d_in[1];
  const float* g5a  = (const float*)d_in[2];
  const float* b5a  = (const float*)d_in[3];
  const float* m5a  = (const float*)d_in[4];
  const float* v5a  = (const float*)d_in[5];
  const float* w5c  = (const float*)d_in[6];
  const float* g5c  = (const float*)d_in[7];
  const float* b5c  = (const float*)d_in[8];
  const float* m5c  = (const float*)d_in[9];
  const float* v5c  = (const float*)d_in[10];
  const float* wq   = (const float*)d_in[11];
  const float* bq   = (const float*)d_in[12];
  const float* wk   = (const float*)d_in[13];
  const float* bk   = (const float*)d_in[14];
  const float* wvw  = (const float*)d_in[15];
  const float* bv   = (const float*)d_in[16];
  const float* gpam = (const float*)d_in[17];
  const float* gcam = (const float*)d_in[18];
  const float* w51  = (const float*)d_in[19];
  const float* g51  = (const float*)d_in[20];
  const float* b51  = (const float*)d_in[21];
  const float* m51  = (const float*)d_in[22];
  const float* v51  = (const float*)d_in[23];
  const float* w52  = (const float*)d_in[24];
  const float* g52  = (const float*)d_in[25];
  const float* b52  = (const float*)d_in[26];
  const float* m52  = (const float*)d_in[27];
  const float* v52  = (const float*)d_in[28];
  const float* w6   = (const float*)d_in[29];
  const float* b6   = (const float*)d_in[30];
  const float* w7   = (const float*)d_in[31];
  const float* b7   = (const float*)d_in[32];
  const float* w8   = (const float*)d_in[33];
  const float* b8   = (const float*)d_in[34];

  char*  base = (char*)d_ws;
  size_t off  = 0;
  auto carve = [&](size_t bytes) -> void* {
    void* p = (void*)(base + off);
    off += (bytes + 255) & ~(size_t)255;
    return p;
  };
  auto carveBf = [&](size_t elems) -> unsigned short* {
    return (unsigned short*)carve(elems * 2);
  };

  (void)carve(16384);                          // guard page for halo underrun
  unsigned short* xN      = carveBf(33554432); // [4,4096,2048]
  unsigned short* w5aB    = carveBf(9437184);  // [3,3,512,2048]
  unsigned short* w5cB    = carveBf(9437184);
  unsigned short* w51B    = carveBf(2359296);  // [3,3,512,512]
  unsigned short* w52B    = carveBf(2359296);
  unsigned short* wqB     = carveBf(32768);
  unsigned short* wkB     = carveBf(32768);
  unsigned short* wvB     = carveBf(262144);
  unsigned short* w6B     = carveBf(16384);    // padded 32x512
  unsigned short* w7B     = carveBf(16384);
  unsigned short* w8B     = carveBf(16384);
  unsigned short* feat1N  = carveBf(8388608);  // [4,4096,512]
  unsigned short* feat2N  = carveBf(8388608);  // [4,4096,512]
  unsigned short* feat2C  = carveBf(8388608);  // [4,512,4096]
  unsigned short* saFeatN = carveBf(8388608);  // [4,4096,512]
  unsigned short* scFeatN = carveBf(8388608);
  unsigned short* saConvN = carveBf(8388608);
  unsigned short* scConvN = carveBf(8388608);
  unsigned short* fsumN   = carveBf(8388608);
  unsigned short* qT      = carveBf(1048576);  // [4,4096,64]
  unsigned short* kT      = carveBf(1048576);
  unsigned short* vBuf    = carveBf(8388608);  // [4,512,4096] (C-major)
  unsigned short* attnP   = carveBf(67108864); // [4,4096,4096]
  float*          energyC = (float*)carve((size_t)4 * 512 * 512 * 4);
  unsigned short* attnC   = carveBf(1048576);  // [4,512,512]

  dim3 blk(256);

  // conversions / repacks
  transpose_cvt_kernel<<<8192, blk, 0, stream>>>(x, xN, 2048, 4096, 4);
  repack_w_kernel<<<4096, blk, 0, stream>>>(w5a, w5aB, 512, 2048);
  repack_w_kernel<<<4096, blk, 0, stream>>>(w5c, w5cB, 512, 2048);
  repack_w_kernel<<<2048, blk, 0, stream>>>(w51, w51B, 512, 512);
  repack_w_kernel<<<2048, blk, 0, stream>>>(w52, w52B, 512, 512);
  cvt_bf16_kernel<<<128, blk, 0, stream>>>(wq,  wqB, (size_t)32768);
  cvt_bf16_kernel<<<128, blk, 0, stream>>>(wk,  wkB, (size_t)32768);
  cvt_bf16_kernel<<<512, blk, 0, stream>>>(wvw, wvB, (size_t)262144);
  pad_cvt_kernel<<<64, blk, 0, stream>>>(w6, w6B, 19, 512, 32);
  pad_cvt_kernel<<<64, blk, 0, stream>>>(w7, w7B, 19, 512, 32);
  pad_cvt_kernel<<<64, blk, 0, stream>>>(w8, w8B, 19, 512, 32);

  // conv5a / conv5c: 2048 -> 512, BN+ReLU (N-major out; conv5c also C-major)
  conv3x3_bn_relu_kernel<<<dim3(16, 64, 4), blk, 0, stream>>>(
      xN, w5aB, g5a, b5a, m5a, v5a, feat1N, nullptr, 2048, 512);
  conv3x3_bn_relu_kernel<<<dim3(16, 64, 4), blk, 0, stream>>>(
      xN, w5cB, g5c, b5c, m5c, v5c, feat2N, feat2C, 2048, 512);

  // PAM: q,k N-major (mode 2), v C-major (mode 0)
  gemm_ldsB_kernel<<<dim3(2, 64, 4), blk, 0, stream>>>(
      wqB, 0, feat1N, bq, 512, 64, 64, 2, qT, nullptr, nullptr, nullptr);
  gemm_ldsB_kernel<<<dim3(2, 64, 4), blk, 0, stream>>>(
      wkB, 0, feat1N, bk, 512, 64, 64, 2, kT, nullptr, nullptr, nullptr);
  gemm_ldsB_kernel<<<dim3(16, 64, 4), blk, 0, stream>>>(
      wvB, 0, feat1N, bv, 512, 512, 512, 0, vBuf, nullptr, nullptr, nullptr);

  // energy[n,m] = q.k (K=64) -> softmax -> out = v @ attn^T (N-major out,
  // residual feat1N)
  gemm_tt_kernel<<<dim3(128, 64, 4), blk, 0, stream>>>(
      qT, (size_t)4096 * 64, 64, kT, (size_t)4096 * 64, 64, 64, 0, 0,
      attnP, nullptr, (size_t)4096 * 4096, 4096, nullptr, nullptr);
  softmax_rows_kernel<<<16384, blk, 0, stream>>>(attnP, 4096);
  gemm_tt_kernel<<<dim3(16, 64, 4), blk, 0, stream>>>(
      vBuf, (size_t)512 * 4096, 4096, attnP, (size_t)4096 * 4096, 4096, 4096, 2, 1,
      saFeatN, nullptr, (size_t)4096 * 512, 512, gpam, feat1N);

  // CAM: energy[c,d] = xf@xf^T (C-major input), softmax, apply (N-major)
  gemm_tt_kernel<<<dim3(16, 8, 4), blk, 0, stream>>>(
      feat2C, (size_t)512 * 4096, 4096, feat2C, (size_t)512 * 4096, 4096, 4096, 1, 0,
      nullptr, energyC, (size_t)512 * 512, 512, nullptr, nullptr);
  cam_softmax_kernel<<<2048, blk, 0, stream>>>(energyC, attnC);
  gemm_ldsB_kernel<<<dim3(16, 64, 4), blk, 0, stream>>>(
      attnC, (size_t)512 * 512, feat2N, nullptr, 512, 512, 512, 3,
      scFeatN, nullptr, gcam, feat2N);

  // conv51 / conv52: 512 -> 512, BN+ReLU (N-major in/out)
  conv3x3_bn_relu_kernel<<<dim3(16, 64, 4), blk, 0, stream>>>(
      saFeatN, w51B, g51, b51, m51, v51, saConvN, nullptr, 512, 512);
  conv3x3_bn_relu_kernel<<<dim3(16, 64, 4), blk, 0, stream>>>(
      scFeatN, w52B, g52, b52, m52, v52, scConvN, nullptr, 512, 512);

  // feat_sum and the three 1x1 class heads (f32 out, 19 of 32 rows)
  add_cvt_kernel<<<4096, blk, 0, stream>>>(saConvN, scConvN, fsumN, (size_t)8388608);
  float* outp = (float*)d_out;
  const int OUT1 = 4 * 19 * 4096; // 311296 floats per output tensor
  gemm_ldsB_kernel<<<dim3(1, 64, 4), blk, 0, stream>>>(
      w8B, 0, fsumN, b8, 512, 32, 19, 1, nullptr, outp, nullptr, nullptr);
  gemm_ldsB_kernel<<<dim3(1, 64, 4), blk, 0, stream>>>(
      w6B, 0, saConvN, b6, 512, 32, 19, 1, nullptr, outp + OUT1, nullptr, nullptr);
  gemm_ldsB_kernel<<<dim3(1, 64, 4), blk, 0, stream>>>(
      w7B, 0, scConvN, b7, 512, 32, 19, 1, nullptr, outp + 2 * OUT1, nullptr, nullptr);
}
